// DendriticAttentionCUDA_62182536512343
// MI455X (gfx1250) — matrix operand order repaired and checked by reading, not compiled, not run on hardware
//
#include <hip/hip_runtime.h>

typedef __attribute__((ext_vector_type(16))) _Float16 v16h;
typedef __attribute__((ext_vector_type(8)))  _Float16 v8h;
typedef __attribute__((ext_vector_type(4)))  _Float16 v4h;
typedef __attribute__((ext_vector_type(8)))  float    v8f;
typedef __attribute__((ext_vector_type(4)))  float    v4f;

#define DEV __device__ __forceinline__

namespace {
constexpr int   kB      = 2;
constexpr int   kH      = 16;
constexpr int   kS      = 2048;
constexpr int   kD      = 64;
constexpr int   kBlk    = 64;          // BLOCK_M == BLOCK_N == 64
constexpr float kThresh = 0.3f;
constexpr float kEps    = 1e-6f;
constexpr float kScale  = 0.125f;      // 1/sqrt(D)
constexpr int   kStr    = 72;          // padded LDS row stride (halves); 144B, 16B-aligned
}

DEV v8f wmma_f16(v16h a, v16h b, v8f c) {
  // (neg_a, A, neg_b, B, c_mod, C, reuse_a, reuse_b)
  return __builtin_amdgcn_wmma_f32_16x16x32_f16(false, a, false, b, (short)0, c,
                                                false, false);
}

// A-fragment from global fp32: elements 0..7 <- p[0..7], 8..15 <- p[16..23], scaled
DEV v16h cvtA(const float* p, float sc) {
  v4f f0 = *(const v4f*)(p);
  v4f f1 = *(const v4f*)(p + 4);
  v4f f2 = *(const v4f*)(p + 16);
  v4f f3 = *(const v4f*)(p + 20);
  v16h r;
#pragma unroll
  for (int e = 0; e < 4; ++e) {
    r[e]      = (_Float16)(f0[e] * sc);
    r[4 + e]  = (_Float16)(f1[e] * sc);
    r[8 + e]  = (_Float16)(f2[e] * sc);
    r[12 + e] = (_Float16)(f3[e] * sc);
  }
  return r;
}

// B-fragment from LDS f16: elements 0..7 <- p[0..7], 8..15 <- p[16..23]; 2x ds_load_b128
DEV v16h ldsB(const _Float16* p) {
  v8h u0 = *(const v8h*)(p);
  v8h u1 = *(const v8h*)(p + 16);
  v16h r;
#pragma unroll
  for (int e = 0; e < 8; ++e) { r[e] = u0[e]; r[8 + e] = u1[e]; }
  return r;
}

// A-fragment from LDS f16: elements 0..7 <- p[0..7], 8..15 <- p[16..23]
DEV v16h ldsA(const _Float16* p) {
  v8h u0 = *(const v8h*)(p);
  v8h u1 = *(const v8h*)(p + 16);
  v16h r;
#pragma unroll
  for (int e = 0; e < 8; ++e) { r[e] = u0[e]; r[8 + e] = u1[e]; }
  return r;
}

__global__ __launch_bounds__(128) void
dendritic_attn_wmma(const float* __restrict__ q, const float* __restrict__ k,
                    const float* __restrict__ v, const float* __restrict__ imp,
                    float* __restrict__ out) {
  __shared__ int s_active;
  __shared__ __align__(16) _Float16 s_k[kBlk * kStr];       // K block, f16 row-major
  __shared__ __align__(16) _Float16 s_vt[kD * kStr];        // V block, f16 transposed
  __shared__ __align__(16) _Float16 s_p[4][16 * kStr];      // per-wave P buffer

  const int mb   = blockIdx.x;       // query 64-block
  const int h    = blockIdx.y;
  const int b    = blockIdx.z;
  const int tid  = threadIdx.x;      // 0..127
  const int w    = tid >> 5;         // wave id 0..3 (16 query rows each)
  const int lane = tid & 31;
  const int nl   = lane & 15;        // N column / A row within fragment
  const int kh   = lane >> 4;        // lane-half selector

  // ---- importance gating for this 64-row query block ----
  if (tid == 0) s_active = 0;
  __syncthreads();
  if (tid < kBlk && imp[b * kS + mb * kBlk + tid] > kThresh) s_active = 1;
  __syncthreads();

  const long  bh      = (long)b * kH + h;
  float*      outBase = out + (bh * kS + (long)mb * kBlk) * kD;

  if (!s_active) {
    for (int i = tid; i < kBlk * kD; i += 128) outBase[i] = 0.0f;
    return;
  }

  const int    m0    = mb * kBlk + w * 16;   // first query row of this wave
  const float* qBase = q + (bh * kS + m0) * kD;
  const float* kBase = k + bh * kS * kD;
  const float* vBase = v + bh * kS * kD;

  // Q tile -> A fragments, scaled by importance/sqrt(D), fp16
  const float  qs   = imp[b * kS + m0 + nl] * kScale;
  const float* qrow = qBase + nl * kD;
  const v16h   a0   = cvtA(qrow + kh * 8, qs);        // d = 0..31
  const v16h   a1   = cvtA(qrow + 32 + kh * 8, qs);   // d = 32..63

  v8f acc[4] = {};                   // output, 4 tiles of 16 d-columns
  _Float16* pw = &s_p[w][0];

  // staging assignment: 1024 float4-chunks per 64x64 block, 8 per thread
  const int sKey = tid >> 4;          // base key row (stride 8 over i)
  const int sDc  = (tid & 15) * 4;    // d-chunk start

  for (int nb = 0; nb <= mb; ++nb) {
    const int  nbase = nb * kBlk;
    const bool diag  = (nb == mb);

    if (nb < mb) {                   // prefetch next K/V block (global_prefetch_b8)
      __builtin_prefetch(kBase + (long)(nbase + kBlk + lane * 2) * kD, 0, 1);
      __builtin_prefetch(vBase + (long)(nbase + kBlk + lane * 2) * kD, 0, 1);
    }

    // ---- cooperative staging: K (row-major) and V^T, fp32 -> f16, once per WG ----
#pragma unroll
    for (int i = 0; i < 8; ++i) {
      const int  key = sKey + i * 8;                       // 0..63
      const long go  = (long)(nbase + key) * kD + sDc;
      v4f kf = *(const v4f*)(kBase + go);                  // global_load_b128
      v4f vf = *(const v4f*)(vBase + go);                  // global_load_b128
      v4h kk;
#pragma unroll
      for (int e = 0; e < 4; ++e) kk[e] = (_Float16)kf[e];
      *(v4h*)(&s_k[key * kStr + sDc]) = kk;                // ds_store_b64
#pragma unroll
      for (int e = 0; e < 4; ++e)
        s_vt[(sDc + e) * kStr + key] = (_Float16)vf[e];    // transposed b16 stores
    }
    __syncthreads();

    // ---- S = (imp*Q/sqrt(D)) @ K^T, per 16x16 tile ----
    v8f c[4];
#pragma unroll
    for (int t = 0; t < 4; ++t) {
      if (!diag || t <= w) {         // wave-uniform: skip fully-masked tiles
        const _Float16* kr = &s_k[(t * 16 + nl) * kStr + kh * 8];
        v8f z = {};
        c[t] = wmma_f16(a0, ldsB(kr), z);
        c[t] = wmma_f16(a1, ldsB(kr + 32), c[t]);
      } else {
#pragma unroll
        for (int r = 0; r < 8; ++r) c[t][r] = -1e30f;
      }
    }

    // ---- causal mask inside the diagonal block ----
    if (diag) {
#pragma unroll
      for (int t = 0; t < 4; ++t)
#pragma unroll
        for (int r = 0; r < 8; ++r)
          if (t * 16 + nl > w * 16 + r + 8 * kh) c[t][r] = -1e30f;
    }

    // ---- per-row softmax over this 64-key block ----
    float rmax[8], rinv[8];
#pragma unroll
    for (int r = 0; r < 8; ++r) {
      float m = fmaxf(fmaxf(c[0][r], c[1][r]), fmaxf(c[2][r], c[3][r]));
      m = fmaxf(m, __shfl_xor(m, 1, 32));
      m = fmaxf(m, __shfl_xor(m, 2, 32));
      m = fmaxf(m, __shfl_xor(m, 4, 32));
      m = fmaxf(m, __shfl_xor(m, 8, 32));
      rmax[r] = m;
    }
#pragma unroll
    for (int r = 0; r < 8; ++r) {
      float s = 0.0f;
#pragma unroll
      for (int t = 0; t < 4; ++t) {
        float e = __expf(c[t][r] - rmax[r]);   // masked (-1e30) -> 0
        c[t][r] = e;
        s += e;
      }
      s += __shfl_xor(s, 1, 32);
      s += __shfl_xor(s, 2, 32);
      s += __shfl_xor(s, 4, 32);
      s += __shfl_xor(s, 8, 32);
      rinv[r] = 1.0f / (s + kEps);
    }

    // ---- P (C layout) -> per-wave LDS -> A fragments ----
#pragma unroll
    for (int t = 0; t < 4; ++t)
#pragma unroll
      for (int r = 0; r < 8; ++r)
        pw[(r + 8 * kh) * kStr + t * 16 + nl] = (_Float16)(c[t][r] * rinv[r]);

    asm volatile("s_wait_dscnt 0" ::: "memory");  // cross-lane LDS dependency

    const v16h pa0 = ldsA(pw + nl * kStr + kh * 8);        // keys 0..31
    const v16h pa1 = ldsA(pw + nl * kStr + 32 + kh * 8);   // keys 32..63

    // ---- out += P @ V  (B fragments from transposed V in LDS) ----
#pragma unroll
    for (int dn = 0; dn < 4; ++dn) {
      const _Float16* vr = &s_vt[(dn * 16 + nl) * kStr + kh * 8];
      acc[dn] = wmma_f16(pa0, ldsB(vr), acc[dn]);
      acc[dn] = wmma_f16(pa1, ldsB(vr + 32), acc[dn]);
    }

    __syncthreads();   // K/V LDS buffers reused next key-block
  }

  // ---- store fp32 output (C-fragment layout -> row-major) ----
#pragma unroll
  for (int dn = 0; dn < 4; ++dn)
#pragma unroll
    for (int r = 0; r < 8; ++r)
      outBase[(w * 16 + r + 8 * kh) * kD + dn * 16 + nl] = acc[dn][r];
}

extern "C" void kernel_launch(void* const* d_in, const int* in_sizes, int n_in,
                              void* d_out, int out_size, void* d_ws, size_t ws_size,
                              hipStream_t stream) {
  const float* q   = (const float*)d_in[0];
  const float* k   = (const float*)d_in[1];
  const float* v   = (const float*)d_in[2];
  const float* imp = (const float*)d_in[3];
  float*       out = (float*)d_out;

  dim3 grid(kS / kBlk, kH, kB);   // (32, 16, 2) workgroups
  hipLaunchKernelGGL(dendritic_attn_wmma, grid, dim3(128), 0, stream,
                     q, k, v, imp, out);
}
